// Network_22127671509084
// MI455X (gfx1250) — compile-verified, matrix operand order checked by reading, not run on hardware
//
#include <hip/hip_runtime.h>
#include <hip/hip_bf16.h>

typedef __attribute__((ext_vector_type(16))) __bf16 v16bf;
typedef __attribute__((ext_vector_type(8)))  float  v8f;

#define THETA   10.0f
#define TAU_SR  10.0f
#define K_SRM   100
#define K_REF   32
#define T_LEN   350

// WMMA 16-bit matrix 16x32 K-index map (CDNA5 ISA 7.12.2): element e of the
// v16bf for a lane in half = lane>>4:  K = e + 8*half + (e>=8 ? 8 : 0)
__device__ __forceinline__ int kmap(int e, int half) {
  return e + 8 * half + ((e >= 8) ? 8 : 0);
}

// ---------------------------------------------------------------------------
// Pack fp32 weights [M,K] into bf16 WMMA A-fragments, zero-padding rows >= M.
// Fragment order: ((mtile*Ktiles + ktile)*32 + lane)*16 bf16 (32B contiguous).
// ---------------------------------------------------------------------------
__global__ void pack_w_kernel(const float* __restrict__ W, __bf16* __restrict__ Apack,
                              int M, int K, int Mtiles) {
  int gid = blockIdx.x * blockDim.x + threadIdx.x;
  int Ktiles = K >> 5;
  if (gid >= Mtiles * Ktiles * 32) return;
  int lane  = gid & 31;
  int tile  = gid >> 5;                 // == mtile*Ktiles + ktile
  int ktile = tile % Ktiles;
  int row   = (tile / Ktiles) * 16 + (lane & 15);
  int half  = lane >> 4;
  v16bf frag;
#pragma unroll
  for (int e = 0; e < 16; ++e) {
    int k = (ktile << 5) + kmap(e, half);
    float v = (row < M) ? W[(size_t)row * K + k] : 0.f;
    frag[e] = (__bf16)v;
  }
  *(v16bf*)(Apack + (size_t)gid * 16) = frag;
}

// ---------------------------------------------------------------------------
// Pack fp32 activations [B][K][T] into bf16 WMMA B-fragments:
// layout [b][ttile][ktile][lane][16], zero-padded for t >= T. Branch-free.
// Lanes 0..15 of a fixed element e read 16 consecutive t -> coalesced 64B.
// ---------------------------------------------------------------------------
__global__ void pack_x_kernel(const float* __restrict__ X, __bf16* __restrict__ Bpack,
                              int K, int Ttiles, int nfraglane /* B*Ttiles*Ktiles*32 */) {
  const int T = T_LEN;
  int gid = blockIdx.x * blockDim.x + threadIdx.x;
  if (gid >= nfraglane) return;
  int Ktiles = K >> 5;
  int lane  = gid & 31;
  int frag  = gid >> 5;
  int ktile = frag % Ktiles;
  int rest  = frag / Ktiles;
  int ttile = rest % Ttiles;
  int b     = rest / Ttiles;
  int tt    = lane & 15;
  int half  = lane >> 4;
  int t     = ttile * 16 + tt;
  float scale = (t < T) ? 1.f : 0.f;    // branch-free tail padding
  int  t_ld   = (t < T) ? t : (T - 1);
  const float* xb = X + ((size_t)b * K + (size_t)(ktile << 5)) * T + t_ld;
  v16bf fragv;
#pragma unroll
  for (int e = 0; e < 16; ++e)
    fragv[e] = (__bf16)(xb[(size_t)kmap(e, half) * T] * scale);
  *(v16bf*)(Bpack + (size_t)gid * 16) = fragv;
}

// ---------------------------------------------------------------------------
// Fragment-direct GEMM: Y[b,m,t] = sum_k W[m,k] * X[b,k,t].
// Each wave owns an MT x NT block of 16x16 C tiles. Per K-step:
// 2*NT b128 loads (B) + 2*MT b128 loads (A) + MT*NT v_wmma. All accesses are
// offsets from the __restrict__ kernel args so they lower to global_load_b128
// (no flat path); no LDS, no barriers, no divergence; MT*NT independent
// accumulator chains keep the matrix pipe busy.
// ---------------------------------------------------------------------------
template <int MT, int NT, int KTILES>
__global__ void __launch_bounds__(128)
gemm_frag_kernel(const __bf16* __restrict__ Bpack, const __bf16* __restrict__ Apack,
                 float* __restrict__ Y, int Ttiles, int Mpad) {
  const int tid  = threadIdx.x;
  const int lane = tid & 31;
  const int wave = tid >> 5;
  const int wpb  = blockDim.x >> 5;
  const int mt0  = (blockIdx.y * wpb + wave) * MT;   // first M tile of this wave
  const int tt0  = blockIdx.x * NT;                  // first T tile of this wave
  const int b    = blockIdx.z;
  const int tt   = lane & 15;
  const int half = lane >> 4;

  const size_t tilestride = (size_t)KTILES * 512;    // elems per (M or T) tile panel
  const size_t aoff = (size_t)mt0 * tilestride + (size_t)lane * 16;
  const size_t boff = ((size_t)b * Ttiles + tt0) * tilestride + (size_t)lane * 16;

  v8f c[MT][NT];
#pragma unroll
  for (int i = 0; i < MT; ++i)
#pragma unroll
    for (int j = 0; j < NT; ++j)
      c[i][j] = (v8f){0.f, 0.f, 0.f, 0.f, 0.f, 0.f, 0.f, 0.f};

  for (int kt = 0; kt < KTILES; ++kt) {
    const size_t ko = (size_t)kt * 512;              // 512 elems per K-step panel slice
    v16bf bb[NT];
#pragma unroll
    for (int j = 0; j < NT; ++j) {
      bb[j] = *(const v16bf*)(Bpack + boff + (size_t)j * tilestride + ko);
      __builtin_prefetch((const void*)(Bpack + boff + (size_t)j * tilestride + ko + 8 * 512),
                         0, 3);                      // near-cache prefetch, 8 K-steps ahead
    }
#pragma unroll
    for (int i = 0; i < MT; ++i) {
      v16bf a = *(const v16bf*)(Apack + aoff + (size_t)i * tilestride + ko);
#pragma unroll
      for (int j = 0; j < NT; ++j)
        c[i][j] = __builtin_amdgcn_wmma_f32_16x16x32_bf16(
                    /*neg_a=*/false, a, /*neg_b=*/false, bb[j],
                    /*c_mod=*/(short)0, c[i][j], /*reuse_a=*/false, /*reuse_b=*/false);
    }
  }

#pragma unroll
  for (int j = 0; j < NT; ++j) {
    const int t = (tt0 + j) * 16 + tt;
    if (t < T_LEN) {
#pragma unroll
      for (int i = 0; i < MT; ++i) {
        float* yb = Y + ((size_t)b * Mpad + (size_t)(mt0 + i) * 16) * T_LEN;
#pragma unroll
        for (int v = 0; v < 8; ++v) {   // C layout: VGPR v -> row v + 8*half, col = lane%16
          int m = v + 8 * half;
          yb[(size_t)m * T_LEN + t] = c[i][j][v];
        }
      }
    }
  }
}

// ---------------------------------------------------------------------------
// In-place causal FIR along T with the SLAYER SRM alpha kernel (100 taps).
// One block per (b, channel) row; row + zero history + kernel staged in LDS.
// ---------------------------------------------------------------------------
__global__ void psp_conv_kernel(float* __restrict__ U) {
  const int T = T_LEN;
  __shared__ float buf[K_SRM - 1 + T_LEN];
  __shared__ float srm[K_SRM];
  float* u = U + (size_t)blockIdx.x * T;
  for (int i = threadIdx.x; i < K_SRM - 1; i += blockDim.x) buf[i] = 0.f;
  for (int i = threadIdx.x; i < T; i += blockDim.x)        buf[K_SRM - 1 + i] = u[i];
  for (int j = threadIdx.x; j < K_SRM; j += blockDim.x) {
    float tj = (float)j;                // TS = 1
    srm[j] = (tj / TAU_SR) * __expf(1.f - tj / TAU_SR);
  }
  __syncthreads();
  for (int t = threadIdx.x; t < T; t += blockDim.x) {
    float acc = 0.f;
#pragma unroll 5
    for (int j = 0; j < K_SRM; ++j)
      acc = fmaf(srm[j], buf[K_SRM - 1 + t - j], acc);
    u[t] = acc;
  }
}

// ---------------------------------------------------------------------------
// Sequential refractory thresholding scan (SLAYER spikeFunction forward).
// One thread per neuron; 32-deep refractory state fully in registers.
// refk[i] = -SCALE_REF*THETA * i * exp(1-i)  (tauRef=1, TS=1); rs[31]=0.
// Safe in place (u[t] read before s[t] written).
// ---------------------------------------------------------------------------
__global__ void spike_scan_kernel(const float* __restrict__ U, float* __restrict__ S,
                                  int Mpad, int Mout, int B) {
  const int T = T_LEN;
  int gid = blockIdx.x * blockDim.x + threadIdx.x;
  if (gid >= B * Mpad) return;
  int b = gid / Mpad;
  int m = gid % Mpad;
  const float* u = U + (size_t)gid * T;
  float* sout = (m < Mout) ? (S + ((size_t)b * Mout + m) * T) : nullptr;

  float rs[K_REF];
#pragma unroll
  for (int j = 0; j < K_REF; ++j) {
    float i = (float)(j + 1);
    rs[j] = (j < K_REF - 1) ? (-2.f * THETA) * i * __expf(1.f - i) : 0.f;
  }
  float r[K_REF];
#pragma unroll
  for (int j = 0; j < K_REF; ++j) r[j] = 0.f;

  for (int t = 0; t < T; ++t) {
    float mem = u[t] + r[0];
    float s   = (mem >= THETA) ? 1.f : 0.f;
#pragma unroll
    for (int j = 0; j < K_REF - 1; ++j)
      r[j] = fmaf(s, rs[j], r[j + 1]);  // shift + add refractory response
    r[K_REF - 1] = 0.f;
    if (sout) sout[t] = s;              // /TS with TS==1
  }
}

// ---------------------------------------------------------------------------
extern "C" void kernel_launch(void* const* d_in, const int* in_sizes, int n_in,
                              void* d_out, int out_size, void* d_ws, size_t ws_size,
                              hipStream_t stream) {
  (void)in_sizes; (void)n_in; (void)out_size; (void)ws_size;
  const float* x  = (const float*)d_in[0];   // [32, 2048, 350] binary spikes
  const float* W1 = (const float*)d_in[1];   // [512, 2048]
  const float* W2 = (const float*)d_in[2];   // [10, 512]
  float* out = (float*)d_out;                // [32, 10, 350]

  const int B = 32, NIN = 2048, NHID = 512, NOUT = 10;
  const int M1tiles = NHID / 16;             // 32
  const int K1tiles = NIN / 32;              // 64
  const int M2tiles = 1;                     // NOUT=10 padded to 16
  const int K2tiles = NHID / 32;             // 16
  const int Ttiles  = (T_LEN + 15) / 16;     // 22 (even, NT=2 blocks -> 11)

  // Workspace carve-up (256B aligned); Bpack2 aliases Bpack1 (dead after GEMM1).
  char* ws = (char*)d_ws;
  size_t off = 0;
  auto alloc = [&](size_t bytes) {
    void* p = ws + off;
    off = (off + bytes + 255) & ~(size_t)255;
    return p;
  };
  __bf16* Apack1 = (__bf16*)alloc((size_t)NHID * NIN * 2);
  __bf16* Apack2 = (__bf16*)alloc((size_t)16 * NHID * 2);
  __bf16* Bpack1 = (__bf16*)alloc((size_t)B * Ttiles * K1tiles * 32 * 16 * 2);
  float*  buf1   = (float*)alloc((size_t)B * NHID * T_LEN * 4); // a1pre/a1/s1 in place
  float*  buf2   = (float*)alloc((size_t)B * 16 * T_LEN * 4);   // a2pre/a2   in place
  __bf16* Bpack2 = Bpack1;                                      // alias, smaller

  // Pack weights into WMMA A-fragment order.
  int wtot1 = M1tiles * K1tiles * 32;
  pack_w_kernel<<<(wtot1 + 255) / 256, 256, 0, stream>>>(W1, Apack1, NHID, NIN, M1tiles);
  int wtot2 = M2tiles * K2tiles * 32;
  pack_w_kernel<<<(wtot2 + 255) / 256, 256, 0, stream>>>(W2, Apack2, NOUT, NHID, M2tiles);

  // Layer 1: pack spikes -> fragment GEMM -> FIR -> spike scan (all on buf1).
  int xtot1 = B * Ttiles * K1tiles * 32;
  pack_x_kernel<<<(xtot1 + 255) / 256, 256, 0, stream>>>(x, Bpack1, NIN, Ttiles, xtot1);
  dim3 g1(Ttiles / 2, M1tiles / (4 * 4), B);  // 4 waves/block, 4 Mtiles x 2 Ttiles each
  gemm_frag_kernel<4, 2, 64><<<g1, 128, 0, stream>>>(Bpack1, Apack1, buf1, Ttiles, NHID);
  psp_conv_kernel<<<B * NHID, 128, 0, stream>>>(buf1);
  spike_scan_kernel<<<(B * NHID + 127) / 128, 128, 0, stream>>>(buf1, buf1, NHID, NHID, B);

  // Layer 2: pack s1 -> tiny fragment GEMM (one wave, 1 Mtile x 2 Ttiles) -> FIR -> out.
  int xtot2 = B * Ttiles * K2tiles * 32;
  pack_x_kernel<<<(xtot2 + 255) / 256, 256, 0, stream>>>(buf1, Bpack2, NHID, Ttiles, xtot2);
  dim3 g2(Ttiles / 2, 1, B);
  gemm_frag_kernel<1, 2, 16><<<g2, 32, 0, stream>>>(Bpack2, Apack2, buf2, K2tiles ? Ttiles : Ttiles, 16);
  psp_conv_kernel<<<B * 16, 128, 0, stream>>>(buf2);
  spike_scan_kernel<<<(B * 16 + 127) / 128, 128, 0, stream>>>(buf2, out, 16, NOUT, B);
}
